// PointUNetObj_48670569398457
// MI455X (gfx1250) — matrix operand order repaired and checked by reading, not compiled, not run on hardware
//
#include <hip/hip_runtime.h>
#include <hip/hip_bf16.h>

typedef __attribute__((ext_vector_type(16))) _Float16 v16h;
typedef __attribute__((ext_vector_type(8)))  float    v8f;

#define EPSF 1e-5f
#define LAUNCH1D(KERN, TOTAL, ...) \
  KERN<<<((((unsigned)(TOTAL)) + 255u) / 256u), 256, 0, stream>>>(__VA_ARGS__)

// ---------------- Farthest point sampling: one block per batch ----------------
__global__ void fps_kernel(const float* __restrict__ pts, int N, int stride,
                           int npoint, int* __restrict__ out) {
  const int b = blockIdx.x, tid = threadIdx.x;
  const float* base = pts + (size_t)b * N * stride;
  float ld[16];
  const int nPer = (N + 255) / 256;
  for (int j = 0; j < nPer; ++j) ld[j] = 1e10f;
  __shared__ float sv[256];
  __shared__ int   si[256];
  int far = 0;
  for (int it = 0; it < npoint; ++it) {
    if (tid == 0) out[(size_t)b * npoint + it] = far;
    const float cx = base[(size_t)far * stride + 0];
    const float cy = base[(size_t)far * stride + 1];
    const float cz = base[(size_t)far * stride + 2];
    float best = -1e30f; int bi = 0;
    for (int j = 0; j < nPer; ++j) {
      int n = j * 256 + tid;
      if (n < N) {
        float dx = base[(size_t)n * stride + 0] - cx;
        float dy = base[(size_t)n * stride + 1] - cy;
        float dz = base[(size_t)n * stride + 2] - cz;
        float v = fminf(ld[j], dx * dx + dy * dy + dz * dz);
        ld[j] = v;
        if (v > best) { best = v; bi = n; }
      }
    }
    sv[tid] = best; si[tid] = bi;
    __syncthreads();
    for (int s = 128; s > 0; s >>= 1) {
      if (tid < s) {
        if (sv[tid + s] > sv[tid] ||
            (sv[tid + s] == sv[tid] && si[tid + s] < si[tid])) {
          sv[tid] = sv[tid + s]; si[tid] = si[tid + s];
        }
      }
      __syncthreads();
    }
    far = si[0];
    __syncthreads();
  }
}

// ---------------- Gather: dst[b,s,:] = src[b, idx[b,s], coff:coff+C] ----------------
__global__ void gather_kernel(const float* __restrict__ src, int N, int stride,
                              int coff, int C, const int* __restrict__ idx,
                              float* __restrict__ dst, int B, int S) {
  int t = blockIdx.x * blockDim.x + threadIdx.x;
  if (t >= B * S * C) return;
  int c = t % C, s = (t / C) % S, b = t / (C * S);
  int n = idx[(size_t)b * S + s];
  dst[t] = src[((size_t)b * N + n) * stride + coff + c];
}

// ---------------- Ball query: first K ascending indices with d^2 <= r^2 ----------------
__global__ void ballq_kernel(const float* __restrict__ pts, int N, int stride,
                             const float* __restrict__ q, int S, int K, float r2,
                             int* __restrict__ idx, int B) {
  int t = blockIdx.x * blockDim.x + threadIdx.x;
  if (t >= B * S) return;
  int b = t / S;
  const float* base = pts + (size_t)b * N * stride;
  float qx = q[(size_t)t * 3], qy = q[(size_t)t * 3 + 1], qz = q[(size_t)t * 3 + 2];
  int* o = idx + (size_t)t * K;
  int cnt = 0;
  for (int n = 0; n < N && cnt < K; ++n) {
    float dx = base[(size_t)n * stride] - qx;
    float dy = base[(size_t)n * stride + 1] - qy;
    float dz = base[(size_t)n * stride + 2] - qz;
    if (dx * dx + dy * dy + dz * dz <= r2) o[cnt++] = n;
  }
  int f = cnt > 0 ? o[0] : 0;
  for (int k = cnt; k < K; ++k) o[k] = f;
}

// ---------------- Grouping -> f16 activation rows, zero-padded to Kpad ----------------
__global__ void group_kernel(const float* __restrict__ pxyz, int N, int xstride,
                             const float* __restrict__ pfeat, int fstride, int foff,
                             int Cfeat, const float* __restrict__ qxyz,
                             const int* __restrict__ idx, _Float16* __restrict__ out,
                             int B, int S, int K, int Kpad) {
  int t = blockIdx.x * blockDim.x + threadIdx.x;
  if (t >= B * S * K) return;
  int n = idx[t];
  int bs = t / K;
  int b = bs / S;
  const float* pp = pxyz + ((size_t)b * N + n) * xstride;
  const float* qq = qxyz + (size_t)bs * 3;
  const float* ff = pfeat + ((size_t)b * N + n) * fstride + foff;
  _Float16* orow = out + (size_t)t * Kpad;
  orow[0] = (_Float16)(pp[0] - qq[0]);
  orow[1] = (_Float16)(pp[1] - qq[1]);
  orow[2] = (_Float16)(pp[2] - qq[2]);
  int c = 3;
  for (int j = 0; j < Cfeat; ++j) orow[c++] = (_Float16)ff[j];
  for (; c < Kpad; ++c) orow[c] = (_Float16)0.f;
}

// ---------------- f32 (O x Cin) -> f16 (O x Kpad), zero padded ----------------
__global__ void w_to_f16(const float* __restrict__ w, _Float16* __restrict__ o,
                         int O, int Cin, int Kpad) {
  int t = blockIdx.x * blockDim.x + threadIdx.x;
  if (t >= O * Kpad) return;
  int c = t % Kpad, r = t / Kpad;
  o[t] = (c < Cin) ? (_Float16)w[(size_t)r * Cin + c] : (_Float16)0.f;
}

// ---------------- WMMA GEMM: Y[M,O] = X[M,Kpad](f16) * W[O,Kpad]^T + bias ----------------
// Block = 8 waves sharing one 16-column weight tile, async-staged to LDS once
// (global_load_async_to_lds_b128 -> s_wait_asynccnt -> barrier). Each wave computes
// MT stacked 16x16 tiles (64x16 for MT=4): one LDS B-fragment feeds MT independent
// v_wmma_f32_16x16x32_f16 accumulator chains per k-step.
// A layout (ISA 7.12.2): lane<16 holds row M=lane, K halves {0..7,16..23}; lane>=16 the rest.
// B layout: lane&15 = column N; K = 8*(lane>>4)+i (+8 for i>=8), same order as A.
// C/D: vgpr j -> M = j + 8*(lane>>4), N = lane&15.
template <int MT>
__global__ void gemm_wmma(const _Float16* __restrict__ X, const _Float16* __restrict__ W,
                          const float* __restrict__ bias, float* __restrict__ Y,
                          int Kpad, int O, int tilesMs, int tilesO) {
  __shared__ _Float16 sW[16 * 512];                 // 16 KB max (Kpad <= 512)
  const int tn = blockIdx.x % tilesO;
  const int o0 = tn << 4;

  // ---- async-stage the 16 x Kpad weight tile into LDS (whole block cooperates) ----
  {
    const int chunks = Kpad >> 3;                   // 8-half (16B) chunks per row
    const int total = 16 * chunks;
    for (int i = threadIdx.x; i < total; i += 256) {
      int row = i / chunks, ch = i % chunks;
      unsigned long long ga =
          (unsigned long long)(W + (size_t)(o0 + row) * Kpad + (size_t)ch * 8);
      unsigned la = (unsigned)(size_t)(sW + row * Kpad + ch * 8);
      asm volatile("global_load_async_to_lds_b128 %0, %1, off"
                   :: "v"(la), "v"(ga) : "memory");
    }
    asm volatile("s_wait_asynccnt 0x0" ::: "memory");
  }
  __syncthreads();

  const int wave = threadIdx.x >> 5;
  const int ms = (blockIdx.x / tilesO) * 8 + wave;
  if (ms >= tilesMs) return;                        // wave-uniform: EXEC all-ones below
  const int m0 = ms * (16 * MT);
  const int lane = threadIdx.x & 31;
  const int hh = lane >> 4;                         // which K-half this lane holds
  const int r  = lane & 15;                         // A row / B column within tile

  v8f c[MT];
  float cb = bias[o0 + r];
#pragma unroll
  for (int mt = 0; mt < MT; ++mt)
#pragma unroll
    for (int j = 0; j < 8; ++j) c[mt][j] = cb;

  const _Float16* xr[MT];
#pragma unroll
  for (int mt = 0; mt < MT; ++mt)
    xr[mt] = X + (size_t)(m0 + mt * 16 + r) * Kpad + (hh << 3);
  const _Float16* wl = sW + r * Kpad + (hh << 3);

  for (int k0 = 0; k0 < Kpad; k0 += 32) {
    v16h bm;
#pragma unroll
    for (int i = 0; i < 8; ++i) { bm[i] = wl[k0 + i]; bm[i + 8] = wl[k0 + 16 + i]; }      // ds_load from staged tile
#pragma unroll
    for (int mt = 0; mt < MT; ++mt) {
      __builtin_prefetch((const void*)(xr[mt] + k0 + 64), 0, 0);
      v16h a;
#pragma unroll
      for (int i = 0; i < 8; ++i) { a[i] = xr[mt][k0 + i]; a[i + 8] = xr[mt][k0 + 16 + i]; }
      c[mt] = __builtin_amdgcn_wmma_f32_16x16x32_f16(false, a, false, bm,
                                                     (short)0, c[mt], false, false);
    }
  }
#pragma unroll
  for (int mt = 0; mt < MT; ++mt) {
    float* yrow = Y + (size_t)(m0 + mt * 16 + (hh << 3)) * O + o0 + r;
#pragma unroll
    for (int j = 0; j < 8; ++j) yrow[(size_t)j * O] = c[mt][j];
  }
}

// ---------------- BN stats per channel over M rows ----------------
__global__ void bn_stats(const float* __restrict__ Y, int M, int O,
                         float* __restrict__ mean, float* __restrict__ var) {
  int o = blockIdx.x;
  float s = 0.f, s2 = 0.f;
  for (int m = threadIdx.x; m < M; m += blockDim.x) {
    float v = Y[(size_t)m * O + o];
    s += v; s2 += v * v;
  }
  __shared__ float ss[256], sq[256];
  ss[threadIdx.x] = s; sq[threadIdx.x] = s2;
  __syncthreads();
  for (int st = 128; st > 0; st >>= 1) {
    if (threadIdx.x < st) { ss[threadIdx.x] += ss[threadIdx.x + st]; sq[threadIdx.x] += sq[threadIdx.x + st]; }
    __syncthreads();
  }
  if (threadIdx.x == 0) {
    float mu = ss[0] / (float)M;
    mean[o] = mu;
    var[o] = sq[0] / (float)M - mu * mu;
  }
}

// ---------------- BN+ReLU -> padded f16 activations (next-layer input) ----------------
__global__ void bn_relu_to_f16(const float* __restrict__ Y, const float* __restrict__ mean,
                               const float* __restrict__ var, const float* __restrict__ g,
                               const float* __restrict__ be, _Float16* __restrict__ Xo,
                               int M, int O, int Kpad) {
  int t = blockIdx.x * blockDim.x + threadIdx.x;
  if (t >= M * Kpad) return;
  int c = t % Kpad;
  int row = t / Kpad;
  if (c >= O) { Xo[t] = (_Float16)0.f; return; }
  float v = Y[(size_t)row * O + c];
  v = g[c] * (v - mean[c]) * rsqrtf(var[c] + EPSF) + be[c];
  Xo[t] = (_Float16)(v > 0.f ? v : 0.f);
}

// ---------------- BN+ReLU in f32 (in-place capable) ----------------
__global__ void bn_relu_f32(const float* __restrict__ Yin, const float* __restrict__ mean,
                            const float* __restrict__ var, const float* __restrict__ g,
                            const float* __restrict__ be, float* __restrict__ Yout,
                            int M, int O) {
  int t = blockIdx.x * blockDim.x + threadIdx.x;
  if (t >= M * O) return;
  int o = t % O;
  float v = Yin[t];
  v = g[o] * (v - mean[o]) * rsqrtf(var[o] + EPSF) + be[o];
  Yout[t] = v > 0.f ? v : 0.f;
}

// ---------------- Max over K neighbor axis ----------------
__global__ void maxpool_kernel(const float* __restrict__ Y, float* __restrict__ out,
                               int BS, int K, int O) {
  int t = blockIdx.x * blockDim.x + threadIdx.x;
  if (t >= BS * O) return;
  int o = t % O;
  size_t bs = t / O;
  const float* p = Y + (size_t)bs * K * O + o;
  float m = -1e30f;
  for (int k = 0; k < K; ++k) m = fmaxf(m, p[(size_t)k * O]);
  out[t] = m;
}

// ---------------- Copy (B,S,C) rows into (B,rowsPerB,C) at rowOff ----------------
__global__ void copy_rows(const float* __restrict__ src, float* __restrict__ dst,
                          int B, int S, int C, int rowsPerB, int rowOff) {
  int t = blockIdx.x * blockDim.x + threadIdx.x;
  if (t >= B * S * C) return;
  int c = t % C, s = (t / C) % S, b = t / (C * S);
  dst[((size_t)b * rowsPerB + rowOff + s) * C + c] = src[t];
}

// ---------------- random_walk: midpoints with k nearest (stable ties) ----------------
__global__ void random_walk_kernel(const float* __restrict__ x, int B, int N, int C,
                                   int k, float* __restrict__ out,
                                   int rowsPerB, int rowOff) {
  int t = blockIdx.x * blockDim.x + threadIdx.x;
  if (t >= B * N) return;
  int b = t / N, n = t % N;
  const float* xb = x + (size_t)b * N * C;
  const float* xn = xb + (size_t)n * C;
  float d[32];
  for (int m = 0; m < N; ++m) {
    float acc = 0.f;
    for (int c = 0; c < C; ++c) { float df = xn[c] - xb[(size_t)m * C + c]; acc += df * df; }
    d[m] = acc;
  }
  unsigned used = 0;
  for (int j = 0; j < k; ++j) {
    int bi = 0; float bv = 1e30f;
    for (int m = 0; m < N; ++m) {
      if (used & (1u << m)) continue;
      if (d[m] < bv) { bv = d[m]; bi = m; }
    }
    used |= (1u << bi);
    float* o = out + ((size_t)b * rowsPerB + rowOff + n * k + j) * C;
    const float* xm = xb + (size_t)bi * C;
    for (int c = 0; c < C; ++c) o[c] = 0.5f * (xn[c] + xm[c]);
  }
}

// ---------------- Pack concat(xyz3, feat) -> f16 rows padded to Kpad ----------------
__global__ void pack2_f16(const float* __restrict__ xyz3, const float* __restrict__ feat,
                          int Cfeat, _Float16* __restrict__ out, int rows, int Kpad) {
  int t = blockIdx.x * blockDim.x + threadIdx.x;
  if (t >= rows * Kpad) return;
  int c = t % Kpad, row = t / Kpad;
  _Float16 v;
  if (c < 3)              v = (_Float16)xyz3[(size_t)row * 3 + c];
  else if (c < 3 + Cfeat) v = (_Float16)feat[(size_t)row * Cfeat + (c - 3)];
  else                    v = (_Float16)0.f;
  out[t] = v;
}

// ---------------- Final tiny FC: (16,256) @ (42,256)^T + b ----------------
__global__ void fc2_kernel(const float* __restrict__ h, const float* __restrict__ w,
                           const float* __restrict__ b, float* __restrict__ out) {
  int t = blockIdx.x * blockDim.x + threadIdx.x;
  if (t >= 16 * 42) return;
  int r = t / 42, c = t % 42;
  float acc = b[c];
  for (int i = 0; i < 256; ++i) acc += h[(size_t)r * 256 + i] * w[(size_t)c * 256 + i];
  out[t] = acc;
}

extern "C" void kernel_launch(void* const* d_in, const int* in_sizes, int n_in,
                              void* d_out, int out_size, void* d_ws, size_t ws_size,
                              hipStream_t stream) {
  (void)in_sizes; (void)n_in; (void)out_size; (void)ws_size;
  const float* xyz = (const float*)d_in[0];
  auto P = [&](int i) { return (const float*)d_in[i]; };
  const float* fc1_w = P(37); const float* fc1_b = P(38);
  const float* bn1_g = P(39); const float* bn1_b = P(40);
  const float* fc2_w = P(41); const float* fc2_b = P(42);

  // ---- workspace bump allocator ----
  char* ws = (char*)d_ws;
  size_t off = 0;
  auto alloc = [&](size_t bytes) -> void* {
    void* p = ws + off;
    off += (bytes + 255) & ~(size_t)255;
    return p;
  };
  _Float16* Xf16   = (_Float16*)alloc((size_t)524288 * 32 * 2);  // activations (f16, padded)
  float*    Yf     = (float*)   alloc((size_t)524288 * 32 * 4);  // layer outputs (f32)
  _Float16* Wf16   = (_Float16*)alloc((size_t)131072 * 2);       // weights (f16, padded)
  float*    bmean  = (float*)   alloc(512 * 4);
  float*    bvar   = (float*)   alloc(512 * 4);
  int*      idxA   = (int*)     alloc((size_t)16 * 512 * 4);
  int*      idxB   = (int*)     alloc((size_t)16 * 256 * 4);
  int*      idxC   = (int*)     alloc((size_t)16 * 32 * 4);
  int*      ball1  = (int*)     alloc((size_t)16 * 512 * 64 * 4);
  int*      ball2  = (int*)     alloc((size_t)16 * 32 * 64 * 4);
  float*    l1_xyz = (float*)   alloc((size_t)16 * 512 * 3 * 4);
  float*    l1_feat= (float*)   alloc((size_t)16 * 512 * 32 * 4);
  float*    l1s_xyz= (float*)   alloc((size_t)16 * 256 * 3 * 4);
  float*    l1s_feat=(float*)   alloc((size_t)16 * 256 * 32 * 4);
  float*    l2_xyz = (float*)   alloc((size_t)16 * 32 * 3 * 4);
  float*    l2_feat= (float*)   alloc((size_t)16 * 32 * 32 * 4);
  float*    l3_xyz = (float*)   alloc((size_t)16 * 1024 * 3 * 4);
  float*    l3_feat= (float*)   alloc((size_t)16 * 1024 * 32 * 4);
  float*    gfeat  = (float*)   alloc((size_t)16 * 512 * 4);

  // ---- GEMM launcher: MT m-tiles per wave, 8 waves (same o-tile) per block ----
  auto launch_gemm = [&](const _Float16* Xp, const float* bb, float* Yp,
                         int M, int Kpad, int O, int MT) {
    int tilesO = O / 16;
    int tilesMs = M / (16 * MT);
    int blocks = tilesO * ((tilesMs + 7) / 8);
    if (MT == 4)
      gemm_wmma<4><<<blocks, 256, 0, stream>>>(Xp, Wf16, bb, Yp, Kpad, O, tilesMs, tilesO);
    else
      gemm_wmma<1><<<blocks, 256, 0, stream>>>(Xp, Wf16, bb, Yp, Kpad, O, tilesMs, tilesO);
  };

  // ---- shared MLP runner: [gemm -> BN stats -> BN+ReLU (-> f16 | maxpool)] x3 ----
  auto run_mlp = [&](int base, int M, int Cin0, const int* Os, int BS, int poolK,
                     float* poolOut) {
    int Cin = Cin0;
    for (int l = 0; l < 3; ++l) {
      int Oc = Os[l];
      int Kpad = ((Cin + 31) / 32) * 32;
      const float* w  = P(base + 4 * l + 0);
      const float* bb = P(base + 4 * l + 1);
      const float* g  = P(base + 4 * l + 2);
      const float* be = P(base + 4 * l + 3);
      LAUNCH1D(w_to_f16, Oc * Kpad, w, Wf16, Oc, Cin, Kpad);
      launch_gemm(Xf16, bb, Yf, M, Kpad, Oc, 4);
      bn_stats<<<Oc, 256, 0, stream>>>(Yf, M, Oc, bmean, bvar);
      if (l < 2) {
        int KpadN = ((Oc + 31) / 32) * 32;
        LAUNCH1D(bn_relu_to_f16, M * KpadN, Yf, bmean, bvar, g, be, Xf16, M, Oc, KpadN);
      } else {
        LAUNCH1D(bn_relu_f32, M * Oc, Yf, bmean, bvar, g, be, Yf, M, Oc);
        LAUNCH1D(maxpool_kernel, BS * Oc, Yf, poolOut, BS, poolK, Oc);
      }
      Cin = Oc;
    }
  };

  // ---- SA1: N=4096 -> S=512, K=64, MLP [16,16,32] ----
  fps_kernel<<<16, 256, 0, stream>>>(xyz, 4096, 6, 512, idxA);
  LAUNCH1D(gather_kernel, 16 * 512 * 3, xyz, 4096, 6, 0, 3, idxA, l1_xyz, 16, 512);
  LAUNCH1D(ballq_kernel, 16 * 512, xyz, 4096, 6, l1_xyz, 512, 64, 0.2f * 0.2f, ball1, 16);
  LAUNCH1D(group_kernel, 16 * 512 * 64, xyz, 4096, 6, xyz, 6, 3, 3, l1_xyz, ball1,
           Xf16, 16, 512, 64, 32);
  { int Os[3] = {16, 16, 32}; run_mlp(1, 524288, 6, Os, 16 * 512, 64, l1_feat); }

  // ---- pidx = FPS(l1_xyz, 256); subsample ----
  fps_kernel<<<16, 256, 0, stream>>>(l1_xyz, 512, 3, 256, idxB);
  LAUNCH1D(gather_kernel, 16 * 256 * 3, l1_xyz, 512, 3, 0, 3, idxB, l1s_xyz, 16, 256);
  LAUNCH1D(gather_kernel, 16 * 256 * 32, l1_feat, 512, 32, 0, 32, idxB, l1s_feat, 16, 256);

  // ---- SA2: N=256 -> S=32, K=64, Cin=35, MLP [32,32,32] ----
  fps_kernel<<<16, 256, 0, stream>>>(l1s_xyz, 256, 3, 32, idxC);
  LAUNCH1D(gather_kernel, 16 * 32 * 3, l1s_xyz, 256, 3, 0, 3, idxC, l2_xyz, 16, 32);
  LAUNCH1D(ballq_kernel, 16 * 32, l1s_xyz, 256, 3, l2_xyz, 32, 64, 0.5f * 0.5f, ball2, 16);
  LAUNCH1D(group_kernel, 16 * 32 * 64, l1s_xyz, 256, 3, l1s_feat, 32, 0, 32, l2_xyz, ball2,
           Xf16, 16, 32, 64, 64);
  { int Os[3] = {32, 32, 32}; run_mlp(13, 32768, 35, Os, 16 * 32, 64, l2_feat); }

  // ---- random_walk upsample + concat into l3 (B,1024,*) ----
  LAUNCH1D(copy_rows, 16 * 512 * 3, l1_xyz, l3_xyz, 16, 512, 3, 1024, 0);
  LAUNCH1D(copy_rows, 16 * 512 * 32, l1_feat, l3_feat, 16, 512, 32, 1024, 0);
  LAUNCH1D(random_walk_kernel, 16 * 32, l2_xyz, 16, 32, 3, 16, l3_xyz, 1024, 512);
  LAUNCH1D(random_walk_kernel, 16 * 32, l2_feat, 16, 32, 32, 16, l3_feat, 1024, 512);

  // ---- SA3 group_all: (B,1024,35) -> MLP [128,256,512] -> max over 1024 ----
  LAUNCH1D(pack2_f16, 16384 * 64, l3_xyz, l3_feat, 32, Xf16, 16384, 64);
  { int Os[3] = {128, 256, 512}; run_mlp(25, 16384, 35, Os, 16, 1024, gfeat); }

  // ---- FC head: h = gfeat @ fc1_w^T + b; batch-BN; ReLU; @ fc2_w^T + b ----
  LAUNCH1D(w_to_f16, 16 * 512, gfeat, Xf16, 16, 512, 512);   // activations -> f16
  LAUNCH1D(w_to_f16, 256 * 512, fc1_w, Wf16, 256, 512, 512);
  launch_gemm(Xf16, fc1_b, Yf, 16, 512, 256, 1);
  bn_stats<<<256, 256, 0, stream>>>(Yf, 16, 256, bmean, bvar);
  LAUNCH1D(bn_relu_f32, 16 * 256, Yf, bmean, bvar, bn1_g, bn1_b, Yf, 16, 256);
  LAUNCH1D(fc2_kernel, 16 * 42, Yf, fc2_w, fc2_b, (float*)d_out);
}
